// GINEClassifier_15152644620445
// MI455X (gfx1250) — compile-verified, matrix operand order checked by reading, not run on hardware
//
#include <hip/hip_runtime.h>
#include <hip/hip_bf16.h>
#include <stdint.h>
#include <stddef.h>

// ---------------------------------------------------------------------------
// GINE GNN forward for MI455X (gfx1250, wave32, WMMA).
// GEMMs: v_wmma_f32_16x16x32_bf16, A tile staged in LDS (bf16), B pre-packed
// into fragment layout. Edge path: coalesced gather + f32 atomic scatter.
// ---------------------------------------------------------------------------

#define HID 128
#define NGRAPH 256
#define BN_EPS 1e-5f

typedef __attribute__((ext_vector_type(16))) __bf16 v16bf;
typedef __attribute__((ext_vector_type(8)))  __bf16 v8bf;
typedef __attribute__((ext_vector_type(4)))  __bf16 v4bf;
typedef __attribute__((ext_vector_type(8)))  float  v8f;

// ---------------------------------------------------------------------------
// Pack a [K x 128] f32 weight matrix into bf16 WMMA B-fragments.
// Layout: index = (((ntile*KT + kstep)*32) + lane)*16 + i
//   lane = column-within-tile (m = lane&15) + K-half select (khalf = lane>>4)
//   i<8  -> k = kstep*32 +      khalf*8 + i
//   i>=8 -> k = kstep*32 + 16 + khalf*8 + (i-8)
// k >= Kmem is zero-padded (used for the K=16 edge encoder).
// ---------------------------------------------------------------------------
__global__ __launch_bounds__(256) void pack_w_kernel(
    const float* __restrict__ W, int Kmem, int KT, __bf16* __restrict__ out)
{
    int idx = blockIdx.x * 256 + threadIdx.x;
    int total = 8 * KT * 32 * 16;
    if (idx >= total) return;
    int i     = idx & 15;
    int lane  = (idx >> 4) & 31;
    int t     = idx >> 9;            // ntile*KT + kstep
    int s     = t % KT;
    int ntile = t / KT;
    int m     = lane & 15;
    int khalf = lane >> 4;
    int within = (i < 8) ? (khalf * 8 + i) : (16 + khalf * 8 + (i - 8));
    int k = s * 32 + within;
    int n = ntile * 16 + m;
    float v = (k < Kmem) ? W[(size_t)k * HID + n] : 0.0f;
    out[idx] = (__bf16)v;
}

// ---------------------------------------------------------------------------
// out[rows x 128] = act( A[rows x Kmem] @ Wpk + bias )
// Block = 256 threads = 8 wave32s; block covers 16 rows x 128 cols,
// wave w computes the 16x16 tile at columns [16w, 16w+16).
// ---------------------------------------------------------------------------
template <int KT, bool RELU, bool OUT_BF16>
__global__ __launch_bounds__(256) void wmma_gemm_kernel(
    const float* __restrict__ A, int rows, int Kmem,
    const __bf16* __restrict__ Wpk, const float* __restrict__ bias,
    void* __restrict__ outv)
{
    constexpr int K = KT * 32;
    __shared__ __attribute__((aligned(16))) __bf16 As[16 * K];

    const int tid     = threadIdx.x;
    const int rowBase = blockIdx.x * 16;

    // Stage the 16 x K A tile into LDS as bf16 (zero-pad K beyond Kmem).
    for (int idx = tid; idx < 16 * K; idx += 256) {
        int r = idx / K;
        int c = idx - r * K;
        int gr = rowBase + r;
        if (gr >= rows) gr = rows - 1;            // clamp (dup row, never stored)
        float v = (c < Kmem) ? A[(size_t)gr * Kmem + c] : 0.0f;
        As[idx] = (__bf16)v;
    }
    __syncthreads();

    const int lane  = tid & 31;
    const int wave  = tid >> 5;                   // N-tile 0..7
    const int m     = lane & 15;
    const int khalf = lane >> 4;
    const int n     = wave * 16 + m;              // global column

    v8f acc;
    const float bv = bias[n];
#pragma unroll
    for (int vi = 0; vi < 8; ++vi) acc[vi] = bv;  // C init = bias broadcast

#pragma unroll
    for (int s = 0; s < KT; ++s) {
        // A fragment: row m, K = s*32 + khalf*8 + {0..7} and +16 + {0..7}
        const __bf16* ap = &As[m * K + s * 32 + khalf * 8];
        v8bf lo = *(const v8bf*)(ap);
        v8bf hi = *(const v8bf*)(ap + 16);
        v16bf afrag = __builtin_shufflevector(lo, hi, 0, 1, 2, 3, 4, 5, 6, 7,
                                              8, 9, 10, 11, 12, 13, 14, 15);
        // B fragment: one contiguous 32B load from the packed weights.
        v16bf bfrag = *(const v16bf*)(Wpk + (((size_t)wave * KT + s) * 32 + lane) * 16);
        acc = __builtin_amdgcn_wmma_f32_16x16x32_bf16(
            false, afrag, false, bfrag, (short)0, acc, false, false);
    }

    if (RELU) {
#pragma unroll
        for (int vi = 0; vi < 8; ++vi) acc[vi] = fmaxf(acc[vi], 0.0f);
    }

    // D layout: VGPR vi -> row = rowBase + vi + khalf*8, col = n.
    // Fast path (rows % 16 == 0 for every launch here): one 64-bit base,
    // 8 straight-line stores with immediate offsets — no per-store branching.
    const int rowOff = rowBase + (khalf << 3);
    if (rowOff + 8 <= rows) {
        if (OUT_BF16) {
            __bf16* op = (__bf16*)outv + (size_t)rowOff * HID + n;
#pragma unroll
            for (int vi = 0; vi < 8; ++vi) op[(size_t)vi * HID] = (__bf16)acc[vi];
        } else {
            float* op = (float*)outv + (size_t)rowOff * HID + n;
#pragma unroll
            for (int vi = 0; vi < 8; ++vi) op[(size_t)vi * HID] = acc[vi];
        }
    } else {
        // Rare ragged tail: predicated per-row stores.
#pragma unroll
        for (int vi = 0; vi < 8; ++vi) {
            int r = rowOff + vi;
            if (r < rows) {
                if (OUT_BF16) ((__bf16*)outv)[(size_t)r * HID + n] = (__bf16)acc[vi];
                else          ((float*)outv)[(size_t)r * HID + n]  = acc[vi];
            }
        }
    }
}

// ---------------------------------------------------------------------------
// Edge message + scatter: agg[dst] += relu(h[src] + ea). 32 lanes per edge,
// 4 channels/lane -> fully coalesced gather of the 512B h row, f32 atomics.
// ---------------------------------------------------------------------------
__global__ __launch_bounds__(256) void message_scatter_kernel(
    const float* __restrict__ h, const __bf16* __restrict__ ea,
    const int* __restrict__ src, const int* __restrict__ dst,
    float* __restrict__ agg, int nEdges)
{
    int idx = blockIdx.x * 256 + threadIdx.x;
    int e = idx >> 5;
    if (e >= nEdges) return;
    int c = (idx & 31) << 2;
    int s = src[e];
    int d = dst[e];
    float4 hv = *(const float4*)(h + (size_t)s * HID + c);
    v4bf  ev = *(const v4bf*)(ea + (size_t)e * HID + c);
    float m0 = fmaxf(hv.x + (float)ev[0], 0.0f);
    float m1 = fmaxf(hv.y + (float)ev[1], 0.0f);
    float m2 = fmaxf(hv.z + (float)ev[2], 0.0f);
    float m3 = fmaxf(hv.w + (float)ev[3], 0.0f);
    float* ap = agg + (size_t)d * HID + c;
    atomicAdd(ap + 0, m0);
    atomicAdd(ap + 1, m1);
    atomicAdd(ap + 2, m2);
    atomicAdd(ap + 3, m3);
}

// global_add_pool: g[batch[n]] += h[n]
__global__ __launch_bounds__(256) void pool_scatter_kernel(
    const float* __restrict__ h, const int* __restrict__ batch,
    float* __restrict__ g, int nNodes)
{
    int idx = blockIdx.x * 256 + threadIdx.x;
    int node = idx >> 5;
    if (node >= nNodes) return;
    int c = (idx & 31) << 2;
    int b = batch[node];
    float4 hv = *(const float4*)(h + (size_t)node * HID + c);
    float* gp = g + (size_t)b * HID + c;
    atomicAdd(gp + 0, hv.x);
    atomicAdd(gp + 1, hv.y);
    atomicAdd(gp + 2, hv.z);
    atomicAdd(gp + 3, hv.w);
}

// BN stats: per-block partial sum / sumsq per channel, atomic combine.
__global__ __launch_bounds__(256) void bn_stats_kernel(
    const float* __restrict__ z, int rows,
    float* __restrict__ sums, float* __restrict__ sumsq)
{
    int c    = threadIdx.x & 127;
    int half = threadIdx.x >> 7;
    int rowsPerBlk = (rows + gridDim.x - 1) / (int)gridDim.x;
    int r0 = blockIdx.x * rowsPerBlk;
    int r1 = r0 + rowsPerBlk; if (r1 > rows) r1 = rows;
    float s = 0.0f, q = 0.0f;
    for (int r = r0 + half; r < r1; r += 2) {
        float v = z[(size_t)r * HID + c];
        s += v;
        q += v * v;
    }
    atomicAdd(&sums[c], s);
    atomicAdd(&sumsq[c], q);
}

__global__ void bn_finalize_kernel(
    const float* __restrict__ sums, const float* __restrict__ sumsq,
    const float* __restrict__ gamma, const float* __restrict__ beta,
    float invN, float* __restrict__ scale, float* __restrict__ shift)
{
    int c = threadIdx.x;                // 128 threads
    float mean = sums[c] * invN;
    float var  = sumsq[c] * invN - mean * mean;   // population var (ddof=0)
    float sc   = gamma[c] * rsqrtf(var + BN_EPS);
    scale[c] = sc;
    shift[c] = beta[c] - mean * sc;
}

__global__ __launch_bounds__(256) void bn_apply_relu_kernel(
    const float* __restrict__ z, const float* __restrict__ scale,
    const float* __restrict__ shift, float* __restrict__ h, int n)
{
    int idx = blockIdx.x * 256 + threadIdx.x;
    if (idx >= n) return;
    int c = idx & 127;
    h[idx] = fmaxf(z[idx] * scale[c] + shift[c], 0.0f);
}

__global__ __launch_bounds__(256) void zero_f32_kernel(float* __restrict__ p, int n)
{
    int i = blockIdx.x * 256 + threadIdx.x;
    if (i < n) p[i] = 0.0f;
}

__global__ __launch_bounds__(256) void copy_f32x4_kernel(
    float4* __restrict__ dst, const float4* __restrict__ src, int n4)
{
    int i = blockIdx.x * 256 + threadIdx.x;
    if (i < n4) dst[i] = src[i];
}

// Final tiny head GEMM: out[256 x 2] = t[256 x 128] @ w[128 x 2] + b
__global__ __launch_bounds__(256) void head2_kernel(
    const float* __restrict__ t, const float* __restrict__ w,
    const float* __restrict__ b, float* __restrict__ out, int rows)
{
    int idx = blockIdx.x * 256 + threadIdx.x;
    if (idx >= rows * 2) return;
    int r = idx >> 1, cls = idx & 1;
    float acc = b[cls];
#pragma unroll 8
    for (int k = 0; k < HID; ++k)
        acc += t[(size_t)r * HID + k] * w[k * 2 + cls];
    out[idx] = acc;
}

// ---------------------------------------------------------------------------
extern "C" void kernel_launch(void* const* d_in, const int* in_sizes, int n_in,
                              void* d_out, int out_size, void* d_ws, size_t ws_size,
                              hipStream_t stream) {
    const float* x          = (const float*)d_in[0];
    const int*   edge_index = (const int*)d_in[1];
    const float* edge_attr  = (const float*)d_in[2];
    const int*   batch      = (const int*)d_in[3];
    const float* node_w     = (const float*)d_in[4];
    const float* node_b     = (const float*)d_in[5];
    const float* edge_w     = (const float*)d_in[6];
    const float* edge_b     = (const float*)d_in[7];
    const float* mlp_w1     = (const float*)d_in[8];
    const float* mlp_b1     = (const float*)d_in[9];
    const float* mlp_w2     = (const float*)d_in[10];
    const float* mlp_b2     = (const float*)d_in[11];
    const float* bn_gamma   = (const float*)d_in[12];
    const float* bn_beta    = (const float*)d_in[13];
    const float* head_w1    = (const float*)d_in[14];
    const float* head_b1    = (const float*)d_in[15];
    const float* head_w2    = (const float*)d_in[16];
    const float* head_b2    = (const float*)d_in[17];
    float* out = (float*)d_out;

    const int NODE_DIM = 64, EDGE_DIM = 16, N_LAYERS = 3;
    const int nNodes = in_sizes[0] / NODE_DIM;
    const int nEdges = in_sizes[2] / EDGE_DIM;
    const int* src = edge_index;            // edge_index[0]
    const int* dst = edge_index + nEdges;   // edge_index[1]

    // ---- carve workspace --------------------------------------------------
    char* p = (char*)d_ws;
    auto carve = [&](size_t bytes) -> void* {
        void* r = (void*)p;
        p += (bytes + 255) & ~(size_t)255;
        return r;
    };
    __bf16* ea    = (__bf16*)carve((size_t)nEdges * HID * sizeof(__bf16));
    float*  h     = (float*) carve((size_t)nNodes * HID * sizeof(float));
    float*  agg   = (float*) carve((size_t)nNodes * HID * sizeof(float));
    float*  z1    = (float*) carve((size_t)nNodes * HID * sizeof(float));
    float*  z2    = (float*) carve((size_t)nNodes * HID * sizeof(float));
    float*  g     = (float*) carve((size_t)NGRAPH * HID * sizeof(float));
    float*  thid  = (float*) carve((size_t)NGRAPH * HID * sizeof(float));
    float*  sums  = (float*) carve(HID * sizeof(float));
    float*  sumsq = (float*) carve(HID * sizeof(float));
    float*  scale = (float*) carve(HID * sizeof(float));
    float*  shift = (float*) carve(HID * sizeof(float));
    __bf16* pk_node = (__bf16*)carve((size_t)8 * 2 * 32 * 16 * sizeof(__bf16)); // KT=2
    __bf16* pk_edge = (__bf16*)carve((size_t)8 * 1 * 32 * 16 * sizeof(__bf16)); // KT=1
    __bf16* pk_w1[N_LAYERS];
    __bf16* pk_w2[N_LAYERS];
    for (int l = 0; l < N_LAYERS; ++l)
        pk_w1[l] = (__bf16*)carve((size_t)8 * 4 * 32 * 16 * sizeof(__bf16));    // KT=4
    for (int l = 0; l < N_LAYERS; ++l)
        pk_w2[l] = (__bf16*)carve((size_t)8 * 4 * 32 * 16 * sizeof(__bf16));
    __bf16* pk_head = (__bf16*)carve((size_t)8 * 4 * 32 * 16 * sizeof(__bf16));
    (void)ws_size; (void)n_in; (void)out_size;

    // ---- pack weights into WMMA B-fragment layout -------------------------
    pack_w_kernel<<<32, 256, 0, stream>>>(node_w, 64, 2, pk_node);
    pack_w_kernel<<<16, 256, 0, stream>>>(edge_w, 16, 1, pk_edge);
    for (int l = 0; l < N_LAYERS; ++l) {
        pack_w_kernel<<<64, 256, 0, stream>>>(mlp_w1 + (size_t)l * HID * HID, HID, 4, pk_w1[l]);
        pack_w_kernel<<<64, 256, 0, stream>>>(mlp_w2 + (size_t)l * HID * HID, HID, 4, pk_w2[l]);
    }
    pack_w_kernel<<<64, 256, 0, stream>>>(head_w1, HID, 4, pk_head);

    const int nodeTiles = (nNodes + 15) / 16;
    const int edgeTiles = (nEdges + 15) / 16;

    // ---- encoders ---------------------------------------------------------
    // h = x @ node_w + node_b   (fp32 out)
    wmma_gemm_kernel<2, false, false><<<nodeTiles, 256, 0, stream>>>(
        x, nNodes, 64, pk_node, node_b, (void*)h);
    // ea = edge_attr @ edge_w + edge_b   (bf16 out, K=16 zero-padded to 32)
    wmma_gemm_kernel<1, false, true><<<edgeTiles, 256, 0, stream>>>(
        edge_attr, nEdges, 16, pk_edge, edge_b, (void*)ea);

    const int nNodeElems = nNodes * HID;
    const int edgeThreads = nEdges * 32;
    const int nodeThreads = nNodes * 32;

    // ---- GINE layers ------------------------------------------------------
    for (int l = 0; l < N_LAYERS; ++l) {
        // agg = h  (so scatter-add directly yields h + segment_sum(msg))
        copy_f32x4_kernel<<<(nNodeElems / 4 + 255) / 256, 256, 0, stream>>>(
            (float4*)agg, (const float4*)h, nNodeElems / 4);
        // agg[dst] += relu(h[src] + ea)
        message_scatter_kernel<<<(edgeThreads + 255) / 256, 256, 0, stream>>>(
            h, ea, src, dst, agg, nEdges);
        // z1 = relu(agg @ w1 + b1)
        wmma_gemm_kernel<4, true, false><<<nodeTiles, 256, 0, stream>>>(
            agg, nNodes, HID, pk_w1[l], mlp_b1 + (size_t)l * HID, (void*)z1);
        // z2 = z1 @ w2 + b2
        wmma_gemm_kernel<4, false, false><<<nodeTiles, 256, 0, stream>>>(
            z1, nNodes, HID, pk_w2[l], mlp_b2 + (size_t)l * HID, (void*)z2);
        // batchnorm (batch statistics) + relu -> h
        zero_f32_kernel<<<1, 256, 0, stream>>>(sums, HID);
        zero_f32_kernel<<<1, 256, 0, stream>>>(sumsq, HID);
        bn_stats_kernel<<<256, 256, 0, stream>>>(z2, nNodes, sums, sumsq);
        bn_finalize_kernel<<<1, HID, 0, stream>>>(
            sums, sumsq, bn_gamma + (size_t)l * HID, bn_beta + (size_t)l * HID,
            1.0f / (float)nNodes, scale, shift);
        bn_apply_relu_kernel<<<(nNodeElems + 255) / 256, 256, 0, stream>>>(
            z2, scale, shift, h, nNodeElems);
    }

    // ---- pooling + head ---------------------------------------------------
    zero_f32_kernel<<<(NGRAPH * HID + 255) / 256, 256, 0, stream>>>(g, NGRAPH * HID);
    pool_scatter_kernel<<<(nodeThreads + 255) / 256, 256, 0, stream>>>(h, batch, g, nNodes);
    // thid = relu(g @ head_w1 + head_b1)
    wmma_gemm_kernel<4, true, false><<<NGRAPH / 16, 256, 0, stream>>>(
        g, NGRAPH, HID, pk_head, head_b1, (void*)thid);
    // out = thid @ head_w2 + head_b2
    head2_kernel<<<(NGRAPH * 2 + 255) / 256, 256, 0, stream>>>(
        thid, head_w2, head_b2, out, NGRAPH);
}
